// GATGeom_16303695856289
// MI455X (gfx1250) — matrix operand order repaired and checked by reading, not compiled, not run on hardware
//
#include <hip/hip_runtime.h>

// ---------------------------------------------------------------------------
// Types for CDNA5 WMMA
// ---------------------------------------------------------------------------
typedef __bf16 bf16_t;
typedef bf16_t bf16x16 __attribute__((ext_vector_type(16)));
typedef float  f32x8   __attribute__((ext_vector_type(8)));

#define TILE_M 128
#define TILE_N 32
#define TILE_K 64
#define LDSS   72   // padded row stride in bf16 elems (72*2=144B, multiple of 16)

// ---------------------------------------------------------------------------
// Helpers
// ---------------------------------------------------------------------------
__device__ __forceinline__ unsigned fkey(float f) {
    unsigned u = __float_as_uint(f);
    return (u & 0x80000000u) ? ~u : (u | 0x80000000u);   // order-preserving key
}
__device__ __forceinline__ float finv(unsigned k) {
    return (k & 0x80000000u) ? __uint_as_float(k & 0x7FFFFFFFu)
                             : __uint_as_float(~k);
}
__device__ __forceinline__ unsigned short f2bf(float f) {
    unsigned u = __float_as_uint(f);
    unsigned r = u + 0x7FFFu + ((u >> 16) & 1u);          // round-nearest-even
    return (unsigned short)(r >> 16);
}

// ---------------------------------------------------------------------------
// Utility kernels
// ---------------------------------------------------------------------------
__global__ void k_fill_zero(unsigned* __restrict__ p, long long n) {
    long long i = blockIdx.x * 256LL + threadIdx.x;
    if (i < n) p[i] = 0u;
}

__global__ void k_cast_bf16(const float* __restrict__ in,
                            unsigned short* __restrict__ out, long long n) {
    long long i = blockIdx.x * 256LL + threadIdx.x;
    if (i < n) out[i] = f2bf(in[i]);
}

// ---------------------------------------------------------------------------
// bf16 WMMA GEMM:  C[M,Ncols] (f32) = A[M,K] (bf16) * B[K,Ncols] (bf16)
// Block: 256 threads (8 waves). Tile: 128 x 32, K staged 64 at a time.
// A tile staged with GLOBAL_LOAD_ASYNC_TO_LDS_B128 (ASYNCcnt-tracked,
// bypasses the register file); B tile transposed into LDS via ds stores.
// ---------------------------------------------------------------------------
__global__ __launch_bounds__(256)
void k_gemm_bf16_wmma(const unsigned short* __restrict__ A,
                      const unsigned short* __restrict__ B,
                      float* __restrict__ C,
                      int M, int K, int Ncols) {
    __shared__ __align__(16) unsigned short As[TILE_M * LDSS];
    __shared__ __align__(16) unsigned short Bs[TILE_N * LDSS];

    const int tid  = threadIdx.x;
    const int lane = tid & 31;
    const int wave = tid >> 5;
    const int m0   = blockIdx.x * TILE_M;
    const int n0   = blockIdx.y * TILE_N;

    const int ar = tid >> 1;             // A-stage row (0..127)
    const int ac = (tid & 1) * 32;       // A-stage col base (bf16 units)
    const int ml = lane & 15;            // row within wave tile / col within n-tile
    const int hi = lane >> 4;            // lane half selector

    f32x8 acc0 = {};
    f32x8 acc1 = {};

    // LDS byte address of this thread's A-stage destination (workgroup-relative
    // offset = low 32 bits of the generic address of a __shared__ object).
    const unsigned ldsA =
        (unsigned)(unsigned long long)(As + ar * LDSS + ac);
    const bool arow_ok = (m0 + ar < M);

    for (int k0 = 0; k0 < K; k0 += TILE_K) {
        // ---- stage A tile: 128 rows x 64 bf16, async global -> LDS ----
        if (arow_ok) {
            unsigned long long gA =
                (unsigned long long)(A + (size_t)(m0 + ar) * K + k0 + ac);
            asm volatile(
                "global_load_async_to_lds_b128 %0, %1, off\n\t"
                "global_load_async_to_lds_b128 %0, %1, off offset:16\n\t"
                "global_load_async_to_lds_b128 %0, %1, off offset:32\n\t"
                "global_load_async_to_lds_b128 %0, %1, off offset:48"
                :: "v"(ldsA), "v"(gA) : "memory");
        } else {
            uint4 z = make_uint4(0, 0, 0, 0);
            uint4* dst = (uint4*)(As + ar * LDSS + ac);
            dst[0] = z; dst[1] = z; dst[2] = z; dst[3] = z;
        }
        // ---- stage B tile transposed: Bs[n][k], 64(k) x 32(n) elems ----
        #pragma unroll
        for (int i = 0; i < 8; ++i) {
            int idx = tid * 8 + i;               // 0..2047
            int kk  = idx >> 5;                  // 0..63
            int nn  = idx & 31;                  // 0..31
            Bs[nn * LDSS + kk] = B[(size_t)(k0 + kk) * Ncols + n0 + nn];
        }
        asm volatile("s_wait_asynccnt 0x0" ::: "memory");
        __syncthreads();

        // ---- two WMMA K-steps over the 64-wide stage ----
        #pragma unroll
        for (int kb = 0; kb < TILE_K; kb += 32) {
            union { bf16x16 v; uint4 q[2]; } aF, bF0, bF1;
            // A fragment (16x32 bf16): lanes 0-15 -> K {0..7,16..23},
            //                          lanes 16-31 -> K {8..15,24..31}
            const unsigned short* abase = As + (wave * 16 + ml) * LDSS + kb;
            aF.q[0] = *(const uint4*)(abase + hi * 8);
            aF.q[1] = *(const uint4*)(abase + 16 + hi * 8);
            // B fragment (32x16 bf16): lanes 0-15 -> K {0..7,16..23} (col n),
            //                          lanes 16-31 -> K {8..15,24..31}
            const unsigned short* b0 = Bs + ml * LDSS + kb + hi * 8;
            bF0.q[0] = *(const uint4*)(b0);
            bF0.q[1] = *(const uint4*)(b0 + 16);
            const unsigned short* b1 = Bs + (16 + ml) * LDSS + kb + hi * 8;
            bF1.q[0] = *(const uint4*)(b1);
            bF1.q[1] = *(const uint4*)(b1 + 16);

            acc0 = __builtin_amdgcn_wmma_f32_16x16x32_bf16(
                false, aF.v, false, bF0.v, (short)0, acc0, false, false);
            acc1 = __builtin_amdgcn_wmma_f32_16x16x32_bf16(
                false, aF.v, false, bF1.v, (short)0, acc1, false, false);
        }
        __syncthreads();
    }

    // ---- store C (16x16 f32 layout: VGPR r -> rows r / r+8 by lane half) ----
    #pragma unroll
    for (int r = 0; r < 8; ++r) {
        int row = m0 + wave * 16 + r + (hi ? 8 : 0);
        if (row < M) {
            size_t base = (size_t)row * Ncols + n0 + ml;
            C[base]      = acc0[r];
            C[base + 16] = acc1[r];
        }
    }
}

// ---------------------------------------------------------------------------
// Attention-score reduction: s_src/s_dst [N,H] from h [N,H*C]
// ---------------------------------------------------------------------------
__global__ void k_scores(const float* __restrict__ h,
                         const float* __restrict__ a_src,
                         const float* __restrict__ a_dst,
                         float* __restrict__ ss, float* __restrict__ sd,
                         long long N, int H, int C) {
    long long idx = blockIdx.x * 256LL + threadIdx.x;
    if (idx >= N * H) return;
    long long n = idx / H;
    int       hh = (int)(idx % H);
    const float* hp = h + n * (size_t)H * C + (size_t)hh * C;
    const float* as = a_src + (size_t)hh * C;
    const float* ad = a_dst + (size_t)hh * C;
    float vs = 0.f, vd = 0.f;
    for (int c = 0; c < C; ++c) { float x = hp[c]; vs += x * as[c]; vd += x * ad[c]; }
    ss[idx] = vs; sd[idx] = vd;
}

// ---------------------------------------------------------------------------
// Edge pass 1: e = leaky_relu(ss[src]+sd[dst]); segment-max via uint keys
// ---------------------------------------------------------------------------
__global__ void k_edge_e_max(const long long* __restrict__ ei,
                             const float* __restrict__ ss,
                             const float* __restrict__ sd,
                             float* __restrict__ e,
                             unsigned* __restrict__ mkey,
                             long long N, long long E, int H) {
    long long idx = blockIdx.x * 256LL + threadIdx.x;
    long long tot = (E + N) * H;
    if (idx >= tot) return;
    long long eid = idx / H;
    int       hh  = (int)(idx % H);
    long long s   = (eid < E) ? ei[eid]     : (eid - E);
    long long d   = (eid < E) ? ei[E + eid] : (eid - E);
    float v = ss[s * H + hh] + sd[d * H + hh];
    v = (v > 0.f) ? v : 0.2f * v;             // leaky_relu, slope 0.2
    e[idx] = v;
    atomicMax(&mkey[d * H + hh], fkey(v));
}

// ---------------------------------------------------------------------------
// Edge pass 2: w = exp(e - m[dst]); denom[dst] += w   (e overwritten with w)
// ---------------------------------------------------------------------------
__global__ void k_edge_w_denom(const long long* __restrict__ ei,
                               float* __restrict__ e,
                               const unsigned* __restrict__ mkey,
                               float* __restrict__ denom,
                               long long N, long long E, int H) {
    long long idx = blockIdx.x * 256LL + threadIdx.x;
    long long tot = (E + N) * H;
    if (idx >= tot) return;
    long long eid = idx / H;
    int       hh  = (int)(idx % H);
    long long d   = (eid < E) ? ei[E + eid] : (eid - E);
    float m = finv(mkey[d * H + hh]);
    float w = expf(e[idx] - m);
    e[idx] = w;
    atomicAdd(&denom[d * H + hh], w);
}

// ---------------------------------------------------------------------------
// Edge pass 3: agg[dst] += (w/denom[dst]) * h[src]   (per head, C feats)
// ---------------------------------------------------------------------------
__global__ void k_edge_agg(const long long* __restrict__ ei,
                           const float* __restrict__ w,
                           const float* __restrict__ denom,
                           const float* __restrict__ h,
                           float* __restrict__ agg,
                           long long N, long long E, int H, int C) {
    long long idx = blockIdx.x * 256LL + threadIdx.x;
    long long tot = (E + N) * H;
    if (idx >= tot) return;
    long long eid = idx / H;
    int       hh  = (int)(idx % H);
    long long s   = (eid < E) ? ei[eid]     : (eid - E);
    long long d   = (eid < E) ? ei[E + eid] : (eid - E);
    const float* hrow = h   + (s * (size_t)H + hh) * C;
    float*       orow = agg + (d * (size_t)H + hh) * C;
    __builtin_prefetch(hrow, 0, 1);           // global_prefetch_b8: line in flight
    float alpha = w[idx] / (denom[d * H + hh] + 1e-16f);
    for (int c = 0; c < C; ++c) atomicAdd(&orow[c], alpha * hrow[c]);
}

// ---------------------------------------------------------------------------
// ELU(agg + bias) -> bf16  (feeds layer-2 GEMM)
// ---------------------------------------------------------------------------
__global__ void k_elu_bias_cast(const float* __restrict__ agg,
                                const float* __restrict__ bias,
                                unsigned short* __restrict__ outb,
                                long long n, int HC) {
    long long i = blockIdx.x * 256LL + threadIdx.x;
    if (i >= n) return;
    float v = agg[i] + bias[i % HC];
    v = (v > 0.f) ? v : expm1f(v);
    outb[i] = f2bf(v);
}

// ---------------------------------------------------------------------------
// Final: out = agg2 + b2 + 1e-6
// ---------------------------------------------------------------------------
__global__ void k_final(const float* __restrict__ agg,
                        const float* __restrict__ bias,
                        float* __restrict__ out, long long n, int C) {
    long long i = blockIdx.x * 256LL + threadIdx.x;
    if (i >= n) return;
    out[i] = agg[i] + bias[i % C] + 1e-6f;
}

// ---------------------------------------------------------------------------
// Host launcher
// ---------------------------------------------------------------------------
static inline unsigned blks(long long n) { return (unsigned)((n + 255) / 256); }

extern "C" void kernel_launch(void* const* d_in, const int* in_sizes, int n_in,
                              void* d_out, int out_size, void* d_ws, size_t ws_size,
                              hipStream_t stream) {
    const float*     x   = (const float*)d_in[0];
    const long long* ei  = (const long long*)d_in[1];
    const float*     W1  = (const float*)d_in[2];
    const float*     as1 = (const float*)d_in[3];
    const float*     ad1 = (const float*)d_in[4];
    const float*     b1  = (const float*)d_in[5];
    const float*     W2  = (const float*)d_in[6];
    const float*     as2 = (const float*)d_in[7];
    const float*     ad2 = (const float*)d_in[8];
    const float*     b2  = (const float*)d_in[9];
    float* out = (float*)d_out;

    const long long FIN = 512, HID = 32, HEADS = 8, NCLASS = 32;
    const long long HC1 = HEADS * HID;                 // 256
    const long long N = in_sizes[0] / FIN;
    const long long E = in_sizes[1] / 2;

    char* ws = (char*)d_ws;
    size_t off = 0;
    auto take = [&](size_t bytes) {
        size_t r = off; off += (bytes + 255) & ~(size_t)255; return r;
    };
    size_t o_xb  = take((size_t)N * FIN * 2);      // x bf16; ALIASED later as agg1 (N*256*4 == N*512*2)
    size_t o_h1  = take((size_t)N * HC1 * 4);      // h1 f32
    size_t o_hb  = take((size_t)N * HC1 * 2);      // ELU(agg1 + b1) bf16
    size_t o_h2  = take((size_t)N * NCLASS * 4);   // h2 f32
    size_t o_ag2 = take((size_t)N * NCLASS * 4);   // agg2 f32
    size_t o_wb  = take((size_t)FIN * HC1 * 2);    // W bf16 (reused for W2)
    size_t o_ss  = take((size_t)N * HEADS * 4);
    size_t o_sd  = take((size_t)N * HEADS * 4);
    size_t o_mk  = take((size_t)N * HEADS * 4);
    size_t o_dn  = take((size_t)N * HEADS * 4);
    size_t o_ew  = take((size_t)(E + N) * HEADS * 4);
    (void)ws_size; (void)n_in; (void)out_size;

    unsigned short* xb  = (unsigned short*)(ws + o_xb);
    float*          ag1 = (float*)(ws + o_xb);     // alias (used only after GEMM1)
    float*          h1  = (float*)(ws + o_h1);
    unsigned short* hb  = (unsigned short*)(ws + o_hb);
    float*          h2  = (float*)(ws + o_h2);
    float*          ag2 = (float*)(ws + o_ag2);
    unsigned short* wb  = (unsigned short*)(ws + o_wb);
    float*          ss  = (float*)(ws + o_ss);
    float*          sd  = (float*)(ws + o_sd);
    unsigned*       mk  = (unsigned*)(ws + o_mk);
    float*          dn  = (float*)(ws + o_dn);
    float*          ew  = (float*)(ws + o_ew);

    const long long EN = E + N;

    // ================= Layer 1 =================
    k_cast_bf16<<<blks(N * FIN), 256, 0, stream>>>(x,  xb, N * FIN);
    k_cast_bf16<<<blks(FIN * HC1), 256, 0, stream>>>(W1, wb, FIN * HC1);

    dim3 g1((unsigned)((N + TILE_M - 1) / TILE_M), (unsigned)(HC1 / TILE_N));
    k_gemm_bf16_wmma<<<g1, 256, 0, stream>>>(xb, wb, h1, (int)N, (int)FIN, (int)HC1);

    // xb is dead now -> reuse region as agg1
    k_fill_zero<<<blks(N * HC1), 256, 0, stream>>>((unsigned*)ag1, N * HC1);
    k_fill_zero<<<blks(N * HEADS), 256, 0, stream>>>(mk, N * HEADS);
    k_fill_zero<<<blks(N * HEADS), 256, 0, stream>>>((unsigned*)dn, N * HEADS);

    k_scores<<<blks(N * HEADS), 256, 0, stream>>>(h1, as1, ad1, ss, sd, N, (int)HEADS, (int)HID);
    k_edge_e_max<<<blks(EN * HEADS), 256, 0, stream>>>(ei, ss, sd, ew, mk, N, E, (int)HEADS);
    k_edge_w_denom<<<blks(EN * HEADS), 256, 0, stream>>>(ei, ew, mk, dn, N, E, (int)HEADS);
    k_edge_agg<<<blks(EN * HEADS), 256, 0, stream>>>(ei, ew, dn, h1, ag1, N, E, (int)HEADS, (int)HID);

    k_elu_bias_cast<<<blks(N * HC1), 256, 0, stream>>>(ag1, b1, hb, N * HC1, (int)HC1);

    // ================= Layer 2 =================
    k_cast_bf16<<<blks(HC1 * NCLASS), 256, 0, stream>>>(W2, wb, HC1 * NCLASS);

    dim3 g2((unsigned)((N + TILE_M - 1) / TILE_M), (unsigned)(NCLASS / TILE_N));
    k_gemm_bf16_wmma<<<g2, 256, 0, stream>>>(hb, wb, h2, (int)N, (int)HC1, (int)NCLASS);

    k_fill_zero<<<blks(N * NCLASS), 256, 0, stream>>>((unsigned*)ag2, N * NCLASS);
    k_fill_zero<<<blks(N), 256, 0, stream>>>(mk, N);
    k_fill_zero<<<blks(N), 256, 0, stream>>>((unsigned*)dn, N);

    k_scores<<<blks(N), 256, 0, stream>>>(h2, as2, ad2, ss, sd, N, 1, (int)NCLASS);
    k_edge_e_max<<<blks(EN), 256, 0, stream>>>(ei, ss, sd, ew, mk, N, E, 1);
    k_edge_w_denom<<<blks(EN), 256, 0, stream>>>(ei, ew, mk, dn, N, E, 1);
    k_edge_agg<<<blks(EN), 256, 0, stream>>>(ei, ew, dn, h2, ag2, N, E, 1, (int)NCLASS);

    k_final<<<blks(N * NCLASS), 256, 0, stream>>>(ag2, b2, out, N * NCLASS, (int)NCLASS);
}
